// AttentionGraphAggregator_27487790695244
// MI455X (gfx1250) — matrix operand order repaired and checked by reading, not compiled, not run on hardware
//
#include <hip/hip_runtime.h>
#include <hip/hip_bf16.h>

#define NN 500000
#define DD 256
#define HH 8
#define DH 32
#define GG 8192

typedef __attribute__((ext_vector_type(16))) _Float16 v16h;
typedef __attribute__((ext_vector_type(8)))  float    v8f;

// ---------------------------------------------------------------------------
// K0: fold graph_query through Wq, then through Wk: qws[h][j] = scale * sum_i
//     q[h,i]*Wk[h*32+i, j];  qlb[h] = scale * q_h . bk_h
// ---------------------------------------------------------------------------
__global__ void k_prep(const float* __restrict__ gq,
                       const float* __restrict__ W,   // in_proj_weight (3D x D)
                       const float* __restrict__ b,   // in_proj_bias  (3D)
                       float* __restrict__ qws, float* __restrict__ qlb)
{
    __shared__ float qv[DD];
    int t = threadIdx.x;                 // 256 threads
    float acc = b[t];                    // bq[t]
    const float* wrow = W + (size_t)t * DD;   // Wq row t
    for (int j = 0; j < DD; ++j) acc += gq[j] * wrow[j];
    qv[t] = acc;
    __syncthreads();

    const float scale = 0.17677669529663688f;   // 1/sqrt(32)
    const float* Wk = W + (size_t)DD * DD;
    const float* bk = b + DD;
    for (int h = 0; h < HH; ++h) {
        float s = 0.f;
        for (int i = 0; i < DH; ++i)
            s += qv[h * DH + i] * Wk[(size_t)(h * DH + i) * DD + t];
        qws[h * DD + t] = s * scale;
    }
    if (t < HH) {
        float s = 0.f;
        for (int i = 0; i < DH; ++i) s += qv[t * DH + i] * bk[t * DH + i];
        qlb[t] = s * scale;
    }
}

// ---------------------------------------------------------------------------
// K0b: one-shot f32 -> f16 conversion of the output projection weight.
// ---------------------------------------------------------------------------
__global__ void k_cvtw(const float* __restrict__ Wo, _Float16* __restrict__ Wo16)
{
    int idx = blockIdx.x * blockDim.x + threadIdx.x;   // DD*DD threads
    Wo16[idx] = (_Float16)Wo[idx];
}

// ---------------------------------------------------------------------------
// K1: segment starts via binary search (graph_idx is sorted)
// ---------------------------------------------------------------------------
__global__ void k_segstart(const int* __restrict__ gidx, int* __restrict__ segstart)
{
    int g = blockIdx.x * blockDim.x + threadIdx.x;
    if (g > GG) return;
    if (g == GG) { segstart[GG] = NN; return; }
    int lo = 0, hi = NN;
    while (lo < hi) { int mid = (lo + hi) >> 1; if (gidx[mid] < g) lo = mid + 1; else hi = mid; }
    segstart[g] = lo;
}

// ---------------------------------------------------------------------------
// K2 (fused): single pass over node_states per graph.
//   For each 32-node LDS chunk: compute logits (wave-per-node, shuffle-
//   reduced), update online-softmax running max/denominator (rescaling the
//   accumulator), then accumulate s[h,:] += e[n,h]*x[n,:] from LDS.
//   Finally attnv16 = f16( Wv.s/denom + bv ).
// ---------------------------------------------------------------------------
__global__ void k_fused(const float* __restrict__ x,
                        const int* __restrict__ segstart,
                        const float* __restrict__ qws, const float* __restrict__ qlb,
                        const float* __restrict__ W, const float* __restrict__ b,
                        _Float16* __restrict__ attnv16)
{
    __shared__ float qs[HH * DD];          // 8 KB   folded Wk^T q (scaled)
    __shared__ float xch[32][260];         // 33 KB  x chunk (float4-aligned pitch)
    __shared__ float lg[32][HH];           // 1 KB   chunk logits
    __shared__ float ech[32][HH];          // 1 KB   chunk exp weights
    __shared__ float slds[HH][DD + 1];     // 8 KB   final s for Wv projection

    int g = blockIdx.x;
    int t = threadIdx.x;
    int s0 = segstart[g], s1 = segstart[g + 1];
    if (s1 == s0) { attnv16[(size_t)g * DD + t] = (_Float16)0.f; return; }  // block-uniform

    for (int i = t; i < HH * DD; i += 256) qs[i] = qws[i];
    float qb[HH];
#pragma unroll
    for (int h = 0; h < HH; ++h) qb[h] = qlb[h];    // uniform scalar loads

    float m[HH], d[HH], acc[HH];
#pragma unroll
    for (int h = 0; h < HH; ++h) { m[h] = -3.0e38f; d[h] = 0.f; acc[h] = 0.f; }

    int wave = t >> 5, lane = t & 31;

    for (int base = s0; base < s1; base += 32) {
        int nn = min(32, s1 - base);
        __syncthreads();                   // previous chunk fully consumed (also qs ready)

        // ---- stage chunk into LDS (float4, coalesced) ----
        {
            const float4* src = (const float4*)(x + (size_t)base * DD);  // nn*64 float4
            int tot = nn * 64;
            for (int idx = t; idx < tot; idx += 256) {
                int r = idx >> 6, q = idx & 63;
                *(float4*)&xch[r][q * 4] = src[r * 64 + q];
            }
        }
        if (base + 32 < s1)                // warm L2 for next chunk (global_prefetch_b8)
            __builtin_prefetch(x + (size_t)(base + 32) * DD + t, 0, 0);
        __syncthreads();

        // ---- logits: wave per node from LDS ----
        for (int r = wave; r < nn; r += 8) {
            float xv[8];
#pragma unroll
            for (int u = 0; u < 8; ++u) xv[u] = xch[r][lane * 8 + u];
            float p[HH];
#pragma unroll
            for (int h = 0; h < HH; ++h) {
                float s = 0.f;
#pragma unroll
                for (int u = 0; u < 8; ++u) s += xv[u] * qs[h * DD + lane * 8 + u];
                p[h] = s;
            }
#pragma unroll
            for (int h = 0; h < HH; ++h) {
                float s = p[h];
                for (int off = 16; off; off >>= 1) s += __shfl_xor(s, off, 32);
                p[h] = s;
            }
            if (lane < HH) lg[r][lane] = p[lane] + qb[lane];
        }
        __syncthreads();

        // ---- online softmax: chunk max (redundant, identical on all threads) ----
        float mn[HH];
#pragma unroll
        for (int h = 0; h < HH; ++h) mn[h] = m[h];
        for (int r = 0; r < nn; ++r)
#pragma unroll
            for (int h = 0; h < HH; ++h) mn[h] = fmaxf(mn[h], lg[r][h]);  // LDS broadcast
#pragma unroll
        for (int h = 0; h < HH; ++h) {
            float sc = __expf(m[h] - mn[h]);    // first chunk: exp(-inf)=0, acc/d are 0
            acc[h] *= sc; d[h] *= sc; m[h] = mn[h];
        }
        // ---- e for chunk: one exp per (node,head) across the block ----
        {
            int i = t >> 3, h = t & 7;
            if (i < nn) ech[i][h] = __expf(lg[i][h] - m[h]);
        }
        __syncthreads();

        // ---- accumulate s and denominator (e read reused for both) ----
        for (int r = 0; r < nn; ++r) {
            float xv = xch[r][t];               // stride-1, conflict-free
#pragma unroll
            for (int h = 0; h < HH; ++h) { float e = ech[r][h]; acc[h] += e * xv; d[h] += e; }
        }
    }
    __syncthreads();
#pragma unroll
    for (int h = 0; h < HH; ++h) slds[h][t] = acc[h];
    __syncthreads();

    // ---- Wv projection: output element t, head t/32 ----
    int ho = t >> 5;
    const float* Wv = W + (size_t)2 * DD * DD;
    const float* bv = b + 2 * DD;
    const float* wr = Wv + (size_t)t * DD;
    float dot = 0.f;
    for (int j = 0; j < DD; ++j) dot += wr[j] * slds[ho][j];   // LDS broadcast per wave
    attnv16[(size_t)g * DD + t] = (_Float16)(dot / d[ho] + bv[t]);
}

// ---------------------------------------------------------------------------
// K3: out = attnv(8192x256) @ Wo^T + bo   via v_wmma_f32_16x16x32_f16
//     All-f16 operands (converted once), f32 accumulation.
//     block = 16 graph rows x full 256 cols; 8 waves x 2 16x16 N-tiles.
//     A tile staged in LDS, pitch 264 halves: per-lane A reads are aligned
//     ds_load_b128 and the 16 M-lanes tile all 64 LDS banks (conflict-free).
// ---------------------------------------------------------------------------
__global__ void k_outproj(const _Float16* __restrict__ attnv16,
                          const _Float16* __restrict__ Wo16,
                          const float* __restrict__ bo, float* __restrict__ out)
{
    __shared__ __align__(16) _Float16 As[16 * 264];
    int t = threadIdx.x;
    size_t gbase = (size_t)blockIdx.x * 16;

    // stage 16x256 f16 A-tile: 512 uint4 (b128) transfers, 2 per thread
    {
        const uint4* src = (const uint4*)(attnv16 + gbase * DD);  // 32 uint4 per row
        uint4* dst = (uint4*)As;                                   // 33 uint4 per padded row
#pragma unroll
        for (int it = 0; it < 2; ++it) {
            int idx = t + it * 256;          // 0..511
            int m = idx >> 5, q = idx & 31;
            dst[m * 33 + q] = src[m * 32 + q];
        }
    }
    __syncthreads();

    int wave = t >> 5, lane = t & 31;
    int mrow  = lane & 15;
    int khalf = lane >> 4;

#pragma unroll
    for (int ti = 0; ti < 2; ++ti) {
        int nb  = (wave + ti * 8) * 16;
        int col = nb + (lane & 15);
        const _Float16* wr = Wo16 + (size_t)col * DD;   // B[k][col] = Wo[col][k]
        v8f c = {0.f, 0.f, 0.f, 0.f, 0.f, 0.f, 0.f, 0.f};
#pragma unroll
        for (int k0 = 0; k0 < DD; k0 += 32) {
            v16h a, bb;
#pragma unroll
            for (int u = 0; u < 16; ++u) {
                // A 16x32 f16 layout: lane-half selects K groups {0-7,16-23}/{8-15,24-31}
                int ka = (u & 7) + khalf * 8 + (u >> 3) * 16;
                a[u] = As[mrow * 264 + k0 + ka];
                // B 32x16 f16 layout: lane-half selects K groups {0-7,16-23}/{8-15,24-31}
                int kb = (u & 7) + khalf * 8 + (u >> 3) * 16;
                bb[u] = wr[k0 + kb];
            }
            c = __builtin_amdgcn_wmma_f32_16x16x32_f16(false, a, false, bb,
                                                       (short)0, c, false, false);
        }
        float bias = bo[col];
#pragma unroll
        for (int r = 0; r < 8; ++r) {
            size_t grow = gbase + r + khalf * 8;   // C/D layout: VGPR r -> M=r / r+8
            out[grow * DD + col] = c[r] + bias;
        }
    }
}

// ---------------------------------------------------------------------------
// K4: count==0 -> 0 ; count==1 -> raw node row
// ---------------------------------------------------------------------------
__global__ void k_fix(const float* __restrict__ x, const int* __restrict__ segstart,
                      float* __restrict__ out)
{
    int g = blockIdx.x;
    int t = threadIdx.x;
    int s0 = segstart[g], s1 = segstart[g + 1];
    int cnt = s1 - s0;
    if (cnt == 0)       out[(size_t)g * DD + t] = 0.f;
    else if (cnt == 1)  out[(size_t)g * DD + t] = x[(size_t)s0 * DD + t];
}

// ---------------------------------------------------------------------------
extern "C" void kernel_launch(void* const* d_in, const int* in_sizes, int n_in,
                              void* d_out, int out_size, void* d_ws, size_t ws_size,
                              hipStream_t stream) {
    const float* x  = (const float*)d_in[0];
    const int* gidx = (const int*)d_in[1];
    // d_in[2] = n_graphs scalar (compile-time GG)
    const float* W  = (const float*)d_in[3];
    const float* b  = (const float*)d_in[4];
    const float* Wo = (const float*)d_in[5];
    const float* bo = (const float*)d_in[6];
    const float* gq = (const float*)d_in[7];
    float* out = (float*)d_out;

    char* ws = (char*)d_ws;
    _Float16* attnv16 = (_Float16*)ws; ws += (size_t)GG * DD * sizeof(_Float16);   // 4 MB
    _Float16* Wo16    = (_Float16*)ws; ws += (size_t)DD * DD * sizeof(_Float16);   // 128 KB
    float* qws        = (float*)ws;    ws += (size_t)HH * DD * sizeof(float);
    float* qlb        = (float*)ws;    ws += 64 * sizeof(float);
    int* segstart     = (int*)ws;      // GG+1 ints

    k_prep    <<<1, 256, 0, stream>>>(gq, W, b, qws, qlb);
    k_cvtw    <<<DD * DD / 256, 256, 0, stream>>>(Wo, Wo16);
    k_segstart<<<(GG + 1 + 255) / 256, 256, 0, stream>>>(gidx, segstart);
    k_fused   <<<GG, 256, 0, stream>>>(x, segstart, qws, qlb, W, b, attnv16);
    k_outproj <<<GG / 16, 256, 0, stream>>>(attnv16, Wo16, bo, out);
    k_fix     <<<GG, 256, 0, stream>>>(x, segstart, out);
}
